// MultiHeadAttention_26259430048704
// MI455X (gfx1250) — matrix operand order repaired and checked by reading, not compiled, not run on hardware
//
#include <hip/hip_runtime.h>
#include <hip/hip_bf16.h>

#define NSEQ 2048
#define HID  1024
#define HEADS 16
#define DK   64

typedef __attribute__((ext_vector_type(16))) __bf16 v16bf;
typedef __attribute__((ext_vector_type(8)))  float  v8f;
typedef unsigned short u16t;

union BF16x16 {
  v16bf v;
  u16t  u[16];
  uint4 q[2];
};

__device__ __forceinline__ u16t f2bf(float f) {
  unsigned int x = __float_as_uint(f);
  x += 0x7fffu + ((x >> 16) & 1u);   // round-to-nearest-even
  return (u16t)(x >> 16);
}

// ---------------------------------------------------------------------------
// Kernel 1a: transpose + convert Wq/Wk/Wv (f32 [k][n]) -> bf16 Wt [n][k]
// ---------------------------------------------------------------------------
__global__ __launch_bounds__(256) void prep_weights(
    const float* __restrict__ Wq, const float* __restrict__ Wk,
    const float* __restrict__ Wv, u16t* __restrict__ WT) {
  int mat = blockIdx.y;
  const float* W = (mat == 0) ? Wq : (mat == 1) ? Wk : Wv;
  u16t* dst = WT + (size_t)mat * HID * HID;
  int idx = blockIdx.x * 256 + threadIdx.x;   // idx = n*HID + k
  int n = idx >> 10;
  int kk = idx & (HID - 1);
  dst[idx] = f2bf(W[(size_t)kk * HID + n]);
}

// ---------------------------------------------------------------------------
// Kernel 1b: convert q/k/v (f32 [n][k]) -> bf16 XB [mat][n][k]
// ---------------------------------------------------------------------------
__global__ __launch_bounds__(256) void prep_x(
    const float* __restrict__ q, const float* __restrict__ k,
    const float* __restrict__ v, u16t* __restrict__ XB) {
  int mat = blockIdx.y;
  const float* X = (mat == 0) ? q : (mat == 1) ? k : v;
  u16t* dst = XB + (size_t)mat * NSEQ * HID;
  int idx = blockIdx.x * 256 + threadIdx.x;
  dst[idx] = f2bf(X[idx]);
}

// ---------------------------------------------------------------------------
// Kernel 2: QKV projection.  One wave computes a 16x64 output tile, with the
// A fragment reused across 4 WMMAs and a software-pipelined double buffer so
// the next k-step's loads are in flight under the current WMMAs.
// ---------------------------------------------------------------------------
__global__ __launch_bounds__(256) void qkv_proj(
    const u16t* __restrict__ XB, const float* __restrict__ bq,
    const float* __restrict__ bk, const float* __restrict__ bv,
    const u16t* __restrict__ WT,
    u16t* __restrict__ QH, u16t* __restrict__ KH, u16t* __restrict__ VT) {
  int wid = threadIdx.x >> 5;
  int lane = threadIdx.x & 31;
  int hf = lane >> 4;          // lane half: selects K sub-group per ISA layouts
  int lr = lane & 15;

  int task = blockIdx.x * 8 + wid;          // 3 * 128 * 16 tasks
  int mat  = task / 2048;                   // 0=q,1=k,2=v
  int rem  = task % 2048;
  int mtile = rem >> 4;                     // 16-row tile
  int ntile = rem & 15;                     // 64-col tile == head index

  const float* Bp = (mat == 0) ? bq : (mat == 1) ? bk : bv;
  const u16t*  Xm = XB + (size_t)mat * NSEQ * HID;
  const u16t*  Wm = WT + (size_t)mat * HID * HID;

  int row = mtile * 16 + lr;                // A row owned by this lane
  const u16t* arow = Xm + (size_t)row * HID + hf * 8;
  const u16t* wbase = Wm + (size_t)(ntile * 64 + lr) * HID + hf * 8;

  v8f acc[4];
#pragma unroll
  for (int t = 0; t < 4; t++) acc[t] = (v8f){0.f,0.f,0.f,0.f,0.f,0.f,0.f,0.f};

  BF16x16 A[2];
  BF16x16 Bm[2][4];

  // prologue: fill buffer 0 with k0 = 0
  A[0].q[0] = *(const uint4*)(arow);
  A[0].q[1] = *(const uint4*)(arow + 16);
#pragma unroll
  for (int t = 0; t < 4; t++) {
    const u16t* wp = wbase + (size_t)t * 16 * HID;
    Bm[0][t].q[0] = *(const uint4*)(wp);
    Bm[0][t].q[1] = *(const uint4*)(wp + 16);
  }

  for (int k0 = 0; k0 < HID - 32; k0 += 32) {
    int cb = (k0 >> 5) & 1;
    int nb = cb ^ 1;
    // issue next k-step's loads before this step's WMMAs
    const u16t* pa = arow + k0 + 32;
    A[nb].q[0] = *(const uint4*)(pa);
    A[nb].q[1] = *(const uint4*)(pa + 16);
#pragma unroll
    for (int t = 0; t < 4; t++) {
      const u16t* wp = wbase + (size_t)t * 16 * HID + k0 + 32;
      Bm[nb][t].q[0] = *(const uint4*)(wp);
      Bm[nb][t].q[1] = *(const uint4*)(wp + 16);
    }
#pragma unroll
    for (int t = 0; t < 4; t++)
      acc[t] = __builtin_amdgcn_wmma_f32_16x16x32_bf16(
          false, A[cb].v, false, Bm[cb][t].v, (short)0, acc[t], false, false);
  }
  {
    int cb = ((HID >> 5) - 1) & 1;          // last buffered step
#pragma unroll
    for (int t = 0; t < 4; t++)
      acc[t] = __builtin_amdgcn_wmma_f32_16x16x32_bf16(
          false, A[cb].v, false, Bm[cb][t].v, (short)0, acc[t], false, false);
  }

  int hh = ntile;                            // head index (64-wide tile)
#pragma unroll
  for (int t = 0; t < 4; t++) {
    int dk = t * 16 + lr;
    float bval = Bp[ntile * 64 + dk];
#pragma unroll
    for (int jj = 0; jj < 8; jj++) {         // D: M = jj + 8*hf
      int n = mtile * 16 + jj + hf * 8;
      float val = acc[t][jj] + bval;
      if (mat == 0) val *= 0.125f;           // dk^-0.5
      u16t bf = f2bf(val);
      if (mat == 0)      QH[((size_t)hh * NSEQ + n) * DK + dk] = bf;
      else if (mat == 1) KH[((size_t)hh * NSEQ + n) * DK + dk] = bf;
      else               VT[((size_t)hh * DK + dk) * NSEQ + n] = bf;
    }
  }
}

// ---------------------------------------------------------------------------
// Kernel 3: fused attention, one wave per (head, 16-query-row tile).
// Online softmax over key chunks of 32.  All global loads for a chunk are
// issued up front so QK WMMAs and softmax overlap the bias/explored stream.
// ---------------------------------------------------------------------------
__global__ __launch_bounds__(128) void attn_kernel(
    const u16t* __restrict__ QH, const u16t* __restrict__ KH,
    const u16t* __restrict__ VT, const float* __restrict__ bias,
    const int* __restrict__ expl, float* __restrict__ out) {
  int wid = threadIdx.x >> 5;
  int lane = threadIdx.x & 31;
  int hf = lane >> 4;
  int lr = lane & 15;

  int task = blockIdx.x * 4 + wid;          // HEADS * 128 tasks
  int h  = task >> 7;
  int mt = task & 127;

  __shared__ __align__(16) u16t lds[4 * 16 * 32];
  u16t* P = lds + wid * 512;                // 16x32 bf16 P tile per wave

  // Load Q A-operand tiles (16x64 split into two 16x32 k-steps), resident.
  int qrow = mt * 16 + lr;
  const u16t* qb = QH + ((size_t)h * NSEQ + qrow) * DK;
  BF16x16 Qa[2];
#pragma unroll
  for (int s = 0; s < 2; s++) {
    Qa[s].q[0] = *(const uint4*)(qb + s * 32 + hf * 8);
    Qa[s].q[1] = *(const uint4*)(qb + s * 32 + 16 + hf * 8);
  }

  float m[8], lsum[8];
  v8f O[4];
#pragma unroll
  for (int jj = 0; jj < 8; jj++) { m[jj] = -1e30f; lsum[jj] = 0.f; }
#pragma unroll
  for (int t = 0; t < 4; t++) O[t] = (v8f){0.f,0.f,0.f,0.f,0.f,0.f,0.f,0.f};

  const float* biasbase = bias + (size_t)h * NSEQ * NSEQ;

  for (int j0 = 0; j0 < NSEQ; j0 += 32) {
    // ================= issue ALL chunk loads up front =================
    // K fragments for two 16-wide key tiles x two 32-deep k-steps
    BF16x16 KB[2][2];
#pragma unroll
    for (int ks = 0; ks < 2; ks++) {
      const u16t* kb0 = KH + ((size_t)h * NSEQ + j0 + lr) * DK + ks * 32 + hf * 8;
      KB[ks][0].q[0] = *(const uint4*)(kb0);
      KB[ks][0].q[1] = *(const uint4*)(kb0 + 16);
      const u16t* kb1 = kb0 + 16 * DK;
      KB[ks][1].q[0] = *(const uint4*)(kb1);
      KB[ks][1].q[1] = *(const uint4*)(kb1 + 16);
    }
    // V fragments (used only after softmax -> long latency tolerance)
    BF16x16 Vb[4];
#pragma unroll
    for (int t = 0; t < 4; t++) {
      const u16t* vb = VT + ((size_t)h * DK + t * 16 + lr) * NSEQ + j0 + hf * 8;
      Vb[t].q[0] = *(const uint4*)(vb);
      Vb[t].q[1] = *(const uint4*)(vb + 16);
    }
    // bias + mask factors (v = kf * (s + b) reproduces reference zeroing)
    int kc0 = j0 + lr;
    int kc1 = j0 + 16 + lr;
    float b0[8], b1[8], kf0[8], kf1[8];
#pragma unroll
    for (int jj = 0; jj < 8; jj++) {
      int qr = mt * 16 + jj + hf * 8;
      const float* br = biasbase + (size_t)qr * NSEQ;
      __builtin_prefetch(br + kc0 + 32, 0, 0);    // next key chunk of bias
      b0[jj] = br[kc0];
      b1[jj] = br[kc1];
      bool keep0 = (qr == 0) || (kc0 == 0) ||
                   (expl[(size_t)(qr - 1) * (NSEQ - 1) + (kc0 - 1)] != 0);
      bool keep1 = (qr == 0) || (kc1 == 0) ||
                   (expl[(size_t)(qr - 1) * (NSEQ - 1) + (kc1 - 1)] != 0);
      kf0[jj] = keep0 ? 1.0f : 0.0f;
      kf1[jj] = keep1 ? 1.0f : 0.0f;
    }

    // ================= S = Q Kh^T =================
    v8f s0 = (v8f){0.f,0.f,0.f,0.f,0.f,0.f,0.f,0.f};
    v8f s1 = (v8f){0.f,0.f,0.f,0.f,0.f,0.f,0.f,0.f};
#pragma unroll
    for (int ks = 0; ks < 2; ks++) {
      s0 = __builtin_amdgcn_wmma_f32_16x16x32_bf16(false, Qa[ks].v, false, KB[ks][0].v,
                                                   (short)0, s0, false, false);
      s1 = __builtin_amdgcn_wmma_f32_16x16x32_bf16(false, Qa[ks].v, false, KB[ks][1].v,
                                                   (short)0, s1, false, false);
    }

    // ================= online softmax =================
    float p0[8], p1[8], scv[8];
#pragma unroll
    for (int jj = 0; jj < 8; jj++) {
      float v0 = kf0[jj] * (s0[jj] + b0[jj]);
      float v1 = kf1[jj] * (s1[jj] + b1[jj]);
      float mx = fmaxf(v0, v1);
#pragma unroll
      for (int off = 1; off < 16; off <<= 1)
        mx = fmaxf(mx, __shfl_xor(mx, off, 32));   // reduce within 16-lane half
      float mn = fmaxf(m[jj], mx);
      float sc = __expf(m[jj] - mn);
      m[jj] = mn;
      float e0 = __expf(v0 - mn);
      float e1 = __expf(v1 - mn);
      float rs = e0 + e1;
#pragma unroll
      for (int off = 1; off < 16; off <<= 1)
        rs += __shfl_xor(rs, off, 32);
      lsum[jj] = lsum[jj] * sc + rs;
      p0[jj] = e0; p1[jj] = e1; scv[jj] = sc;
    }

    // rescale output accumulators by row scale
#pragma unroll
    for (int t = 0; t < 4; t++)
#pragma unroll
      for (int jj = 0; jj < 8; jj++) O[t][jj] *= scv[jj];

    // ---- P (C layout) -> LDS -> A layout ------------------------------
#pragma unroll
    for (int jj = 0; jj < 8; jj++) {
      int Mr = jj + hf * 8;
      P[Mr * 32 + lr]      = f2bf(p0[jj]);
      P[Mr * 32 + 16 + lr] = f2bf(p1[jj]);
    }
    asm volatile("s_wait_dscnt 0" ::: "memory");
    BF16x16 Pa;
    Pa.q[0] = *(const uint4*)(P + lr * 32 + hf * 8);
    Pa.q[1] = *(const uint4*)(P + lr * 32 + 16 + hf * 8);

    // ================= O += P @ V (V fragments already resident) =======
#pragma unroll
    for (int t = 0; t < 4; t++) {
      O[t] = __builtin_amdgcn_wmma_f32_16x16x32_bf16(false, Pa.v, false, Vb[t].v,
                                                     (short)0, O[t], false, false);
    }
  }

  // ---- normalize and write out[n][h*64+dk] -----------------------------
#pragma unroll
  for (int jj = 0; jj < 8; jj++) {
    float inv = 1.0f / lsum[jj];
    int n = mt * 16 + jj + hf * 8;
#pragma unroll
    for (int t = 0; t < 4; t++) {
      out[(size_t)n * HID + h * 64 + t * 16 + lr] = O[t][jj] * inv;
    }
  }
}

// ---------------------------------------------------------------------------
extern "C" void kernel_launch(void* const* d_in, const int* in_sizes, int n_in,
                              void* d_out, int out_size, void* d_ws, size_t ws_size,
                              hipStream_t stream) {
  const float* q         = (const float*)d_in[0];
  const float* k         = (const float*)d_in[1];
  const float* v         = (const float*)d_in[2];
  const float* attn_bias = (const float*)d_in[3];
  const int*   explored  = (const int*)d_in[4];
  const float* Wq        = (const float*)d_in[5];
  const float* bq        = (const float*)d_in[6];
  const float* Wk        = (const float*)d_in[7];
  const float* bk        = (const float*)d_in[8];
  const float* Wv        = (const float*)d_in[9];
  const float* bv        = (const float*)d_in[10];
  float* out = (float*)d_out;

  // workspace layout (bf16):
  //   WT 3*HID*HID (6 MB) | XB 3*NSEQ*HID (12 MB) | QH | KH | VT (4 MB each)
  u16t* WT = (u16t*)d_ws;
  u16t* XB = WT + (size_t)3 * HID * HID;
  u16t* QH = XB + (size_t)3 * NSEQ * HID;
  u16t* KH = QH + (size_t)NSEQ * HID;
  u16t* VT = KH + (size_t)NSEQ * HID;

  dim3 g1(HID * HID / 256, 3);
  prep_weights<<<g1, 256, 0, stream>>>(Wq, Wk, Wv, WT);

  dim3 g2(NSEQ * HID / 256, 3);
  prep_x<<<g2, 256, 0, stream>>>(q, k, v, XB);

  qkv_proj<<<3 * (NSEQ / 16) * (HID / 64) / 8, 256, 0, stream>>>(
      XB, bq, bk, bv, WT, QH, KH, VT);

  attn_kernel<<<HEADS * (NSEQ / 16) / 4, 128, 0, stream>>>(
      QH, KH, VT, attn_bias, explored, out);
}
